// CustomGCNModel_80453327389046
// MI455X (gfx1250) — matrix operand order repaired and checked by reading, not compiled, not run on hardware
//
#include <hip/hip_runtime.h>
#include <hip/hip_bf16.h>
#include <math.h>

typedef __attribute__((ext_vector_type(16))) __bf16          v16bf;
typedef __attribute__((ext_vector_type(8)))  float           v8f;

__device__ __forceinline__ unsigned short f2bf(float f) {
  union { __bf16 b; unsigned short u; } c;
  c.b = (__bf16)f;                           // hardware f32->bf16 (RNE)
  return c.u;
}
__device__ __forceinline__ float bf2f(unsigned short h) {
  return __uint_as_float(((unsigned)h) << 16);
}

// ---------------- degree / norm ----------------
__global__ void k_deg_init(float* __restrict__ deg, int n) {
  int i = blockIdx.x * 256 + threadIdx.x;
  if (i < n) deg[i] = 1.0f;                  // self-loop weight
}
__global__ void k_deg_acc(const int* __restrict__ row, const float* __restrict__ w,
                          float* __restrict__ deg, int e) {
  int i = blockIdx.x * 256 + threadIdx.x;
  if (i < e) atomicAdd(&deg[row[i]], w[i]);
}
__global__ void k_dinv(const float* __restrict__ deg, float* __restrict__ dinv, int n) {
  int i = blockIdx.x * 256 + threadIdx.x;
  if (i < n) { float d = deg[i]; dinv[i] = (d > 0.0f) ? rsqrtf(d) : 0.0f; }
}
__global__ void k_norm(const int* __restrict__ row, const int* __restrict__ col,
                       const float* __restrict__ w, const float* __restrict__ dinv,
                       float* __restrict__ nrm, int e) {
  int i = blockIdx.x * 256 + threadIdx.x;
  if (i < e) nrm[i] = dinv[row[i]] * w[i] * dinv[col[i]];
}

// Pack 8 floats (two float4) into A-fragment elements [base, base+8)
__device__ __forceinline__ void pack8(v16bf& a, int base, float4 p0, float4 p1, bool relu) {
  float v0=p0.x, v1=p0.y, v2=p0.z, v3=p0.w, v4=p1.x, v5=p1.y, v6=p1.z, v7=p1.w;
  if (relu) { v0=fmaxf(v0,0.f); v1=fmaxf(v1,0.f); v2=fmaxf(v2,0.f); v3=fmaxf(v3,0.f);
              v4=fmaxf(v4,0.f); v5=fmaxf(v5,0.f); v6=fmaxf(v6,0.f); v7=fmaxf(v7,0.f); }
  a[base+0]=(__bf16)v0; a[base+1]=(__bf16)v1; a[base+2]=(__bf16)v2; a[base+3]=(__bf16)v3;
  a[base+4]=(__bf16)v4; a[base+5]=(__bf16)v5; a[base+6]=(__bf16)v6; a[base+7]=(__bf16)v7;
}

// ---------------- GEMM1: xw(bf16) = x @ W1, WMMA bf16 ----------------
// LDS holds W1 pre-swizzled into B-fragment order:
//   sB[((t*4 + kc)*32 + lane)*16 + e]  <-  W1[(kc*32 + (lane>=16?8:0) + (e&7) + (e>=8?16:0))*128 + t*16 + (lane&15)]
__global__ void __launch_bounds__(256) k_gemm1(const float* __restrict__ x,
                                               const float* __restrict__ W1,
                                               unsigned short* __restrict__ xw, int n) {
  __shared__ __align__(32) __bf16 sB[8 * 4 * 32 * 16];   // 32KB
  for (int i = threadIdx.x; i < 128 * 128; i += 256) {
    int k = i >> 7, ncol = i & 127;
    int t = ncol >> 4, l16s = ncol & 15;
    int kc = k >> 5, krem = k & 31;
    int lane = ((krem >> 3) & 1) * 16 + l16s;             // half-wave from K octet parity
    int e2 = (krem & 7) + ((krem >> 4) << 3);
    sB[(((t * 4 + kc) * 32 + lane) << 4) + e2] = (__bf16)W1[i];
  }
  __syncthreads();

  const int wave = threadIdx.x >> 5;
  const int lane = threadIdx.x & 31;
  const int hl   = lane >> 4;
  const int l16  = lane & 15;
  const int rowBase = (blockIdx.x * 8 + wave) * 16;
  if (rowBase >= n) return;

  v8f zero = {0.f,0.f,0.f,0.f,0.f,0.f,0.f,0.f};
  v8f acc[8];
  #pragma unroll
  for (int t = 0; t < 8; ++t) acc[t] = zero;

  const int arow = rowBase + l16;
  const int rowc = (arow < n) ? arow : (n - 1);
  const float* xrow = x + (size_t)rowc * 128;
  const v16bf* frag = (const v16bf*)sB;

  for (int kc = 0; kc < 4; ++kc) {
    v16bf a;
    const int kb = kc * 32 + hl * 8;                      // A: 16-bit 16x32 layout
    float4 p0 = *(const float4*)(xrow + kb);
    float4 p1 = *(const float4*)(xrow + kb + 4);
    float4 q0 = *(const float4*)(xrow + kb + 16);
    float4 q1 = *(const float4*)(xrow + kb + 20);
    pack8(a, 0, p0, p1, false);
    pack8(a, 8, q0, q1, false);
    #pragma unroll
    for (int t = 0; t < 8; ++t) {
      v16bf b = frag[(t * 4 + kc) * 32 + lane];           // one 32B contiguous LDS read
      acc[t] = __builtin_amdgcn_wmma_f32_16x16x32_bf16(
          false, a, false, b, (short)0, acc[t], false, false);
    }
  }
  #pragma unroll
  for (int t = 0; t < 8; ++t)
    #pragma unroll
    for (int r = 0; r < 8; ++r) {
      const int row = rowBase + hl * 8 + r;               // C/D: VGPR r -> M=r (+8 upper)
      if (row < n) xw[(size_t)row * 128 + t * 16 + l16] = f2bf(acc[t][r]);
    }
}

// ---------------- GEMM2: h2(f32, 48-pad) = relu(h) @ W2, WMMA bf16 ----------------
__global__ void __launch_bounds__(256) k_gemm2(const float* __restrict__ h,
                                               const float* __restrict__ W2,
                                               float* __restrict__ h2, int n) {
  __shared__ __align__(32) __bf16 sB[3 * 4 * 32 * 16];    // 12KB, W2 padded 40->48
  for (int i = threadIdx.x; i < 128 * 48; i += 256) {
    int k = i / 48, c = i % 48;
    int t = c >> 4, l16s = c & 15;
    int kc = k >> 5, krem = k & 31;
    int lane = ((krem >> 3) & 1) * 16 + l16s;
    int e2 = (krem & 7) + ((krem >> 4) << 3);
    float v = (c < 40) ? W2[k * 40 + c] : 0.0f;
    sB[(((t * 4 + kc) * 32 + lane) << 4) + e2] = (__bf16)v;
  }
  __syncthreads();

  const int wave = threadIdx.x >> 5;
  const int lane = threadIdx.x & 31;
  const int hl   = lane >> 4;
  const int l16  = lane & 15;
  const int rowBase = (blockIdx.x * 8 + wave) * 16;
  if (rowBase >= n) return;

  v8f zero = {0.f,0.f,0.f,0.f,0.f,0.f,0.f,0.f};
  v8f acc[3];
  #pragma unroll
  for (int t = 0; t < 3; ++t) acc[t] = zero;

  const int arow = rowBase + l16;
  const int rowc = (arow < n) ? arow : (n - 1);
  const float* hrow = h + (size_t)rowc * 128;
  const v16bf* frag = (const v16bf*)sB;

  for (int kc = 0; kc < 4; ++kc) {
    v16bf a;
    const int kb = kc * 32 + hl * 8;
    float4 p0 = *(const float4*)(hrow + kb);
    float4 p1 = *(const float4*)(hrow + kb + 4);
    float4 q0 = *(const float4*)(hrow + kb + 16);
    float4 q1 = *(const float4*)(hrow + kb + 20);
    pack8(a, 0, p0, p1, true);                            // fused ReLU
    pack8(a, 8, q0, q1, true);
    #pragma unroll
    for (int t = 0; t < 3; ++t) {
      v16bf b = frag[(t * 4 + kc) * 32 + lane];
      acc[t] = __builtin_amdgcn_wmma_f32_16x16x32_bf16(
          false, a, false, b, (short)0, acc[t], false, false);
    }
  }
  #pragma unroll
  for (int t = 0; t < 3; ++t)
    #pragma unroll
    for (int r = 0; r < 8; ++r) {
      const int row = rowBase + hl * 8 + r;
      if (row < n) h2[(size_t)row * 48 + t * 16 + l16] = acc[t][r];
    }
}

// ---------------- self-loop init + edge scatter ----------------
__global__ void k_self1(const float* __restrict__ dinv, const unsigned short* __restrict__ xw,
                        float* __restrict__ h, int n) {
  long long i = (long long)blockIdx.x * 256 + threadIdx.x;
  if (i >= (long long)n * 128) return;
  int node = (int)(i >> 7);
  float dv = dinv[node];
  h[i] = dv * dv * bf2f(xw[i]);
}
__global__ void k_agg1(const int* __restrict__ row, const int* __restrict__ col,
                       const float* __restrict__ nrm, const unsigned short* __restrict__ xw,
                       float* __restrict__ h, int e) {
  int wid  = (blockIdx.x * 256 + threadIdx.x) >> 5;       // one wave per edge
  int lane = threadIdx.x & 31;
  if (wid >= e) return;
  int r = row[wid], c = col[wid];
  float wv = nrm[wid];
  const unsigned short* s = xw + (size_t)c * 128 + lane * 4;
  float* d = h + (size_t)r * 128 + lane * 4;
  uint2 pk = *(const uint2*)s;                            // 8B gather (L2-resident)
  atomicAdd(d + 0, wv * bf2f((unsigned short)(pk.x & 0xFFFFu)));
  atomicAdd(d + 1, wv * bf2f((unsigned short)(pk.x >> 16)));
  atomicAdd(d + 2, wv * bf2f((unsigned short)(pk.y & 0xFFFFu)));
  atomicAdd(d + 3, wv * bf2f((unsigned short)(pk.y >> 16)));
}
__global__ void k_self2(const float* __restrict__ dinv, const float* __restrict__ h2,
                        float* __restrict__ out, int n) {
  long long i = (long long)blockIdx.x * 256 + threadIdx.x;
  if (i >= (long long)n * 40) return;
  int node = (int)(i / 40), f = (int)(i % 40);
  float dv = dinv[node];
  out[i] = dv * dv * h2[(size_t)node * 48 + f];
}
__global__ void k_agg2(const int* __restrict__ row, const int* __restrict__ col,
                       const float* __restrict__ nrm, const float* __restrict__ h2,
                       float* __restrict__ out, int e) {
  int wid  = (blockIdx.x * 256 + threadIdx.x) >> 5;
  int lane = threadIdx.x & 31;
  if (wid >= e) return;
  int r = row[wid], c = col[wid];
  float wv = nrm[wid];
  const float* s = h2 + (size_t)c * 48;
  float* d = out + (size_t)r * 40;
  atomicAdd(d + lane, wv * s[lane]);
  if (lane < 8) atomicAdd(d + 32 + lane, wv * s[32 + lane]);
}

// ---------------- log-softmax over 40 cols, wave per row ----------------
__global__ void k_lsm(float* __restrict__ out, int n) {
  int wid  = (blockIdx.x * 256 + threadIdx.x) >> 5;
  int lane = threadIdx.x & 31;
  if (wid >= n) return;
  float* p = out + (size_t)wid * 40;
  float v1 = p[lane];
  float v2 = (lane < 8) ? p[32 + lane] : -INFINITY;
  float m = fmaxf(v1, v2);
  for (int o = 16; o > 0; o >>= 1) m = fmaxf(m, __shfl_xor(m, o, 32));
  float s = expf(v1 - m) + ((lane < 8) ? expf(v2 - m) : 0.0f);
  for (int o = 16; o > 0; o >>= 1) s += __shfl_xor(s, o, 32);
  float lse = m + logf(s);
  p[lane] = v1 - lse;
  if (lane < 8) p[32 + lane] = v2 - lse;
}

extern "C" void kernel_launch(void* const* d_in, const int* in_sizes, int n_in,
                              void* d_out, int out_size, void* d_ws, size_t ws_size,
                              hipStream_t stream) {
  const float* x  = (const float*)d_in[0];
  const int*   ei = (const int*)d_in[1];
  const float* ew = (const float*)d_in[2];
  const float* W1 = (const float*)d_in[3];
  const float* W2 = (const float*)d_in[4];
  const int n = in_sizes[0] / 128;
  const int e = in_sizes[2];
  const int* row = ei;            // edge_index[0]
  const int* col = ei + e;        // edge_index[1]
  float* out = (float*)d_out;

  char* p = (char*)d_ws;
  auto carve = [&](size_t bytes) -> char* {
    char* q = p; p += (bytes + 255) & ~(size_t)255; return q;
  };
  float*          deg  = (float*)carve((size_t)n * 4);
  float*          dinv = (float*)carve((size_t)n * 4);
  float*          nrm  = (float*)carve((size_t)e * 4);
  unsigned short* xw   = (unsigned short*)carve((size_t)n * 128 * 2);
  float*          h    = (float*)carve((size_t)n * 128 * 4);
  float*          h2   = (float*)xw;   // xw dead once GEMM2 starts; 48*4 <= 128*2 bytes/row

  const int B = 256;
  k_deg_init<<<(n + B - 1) / B, B, 0, stream>>>(deg, n);
  k_deg_acc <<<(e + B - 1) / B, B, 0, stream>>>(row, ew, deg, e);
  k_dinv    <<<(n + B - 1) / B, B, 0, stream>>>(deg, dinv, n);
  k_norm    <<<(e + B - 1) / B, B, 0, stream>>>(row, col, ew, dinv, nrm, e);

  const int rowsPerBlk = 8 * 16;  // 8 waves x 16 rows
  k_gemm1<<<(n + rowsPerBlk - 1) / rowsPerBlk, B, 0, stream>>>(x, W1, xw, n);

  {
    long long tot = (long long)n * 128;
    k_self1<<<(unsigned)((tot + B - 1) / B), B, 0, stream>>>(dinv, xw, h, n);
  }
  k_agg1<<<(e + 7) / 8, B, 0, stream>>>(row, col, nrm, xw, h, e);   // wave per edge

  k_gemm2<<<(n + rowsPerBlk - 1) / rowsPerBlk, B, 0, stream>>>(h, W2, h2, n);

  {
    long long tot = (long long)n * 40;
    k_self2<<<(unsigned)((tot + B - 1) / B), B, 0, stream>>>(dinv, h2, out, n);
  }
  k_agg2<<<(e + 7) / 8, B, 0, stream>>>(row, col, nrm, h2, out, e);

  k_lsm<<<(n + 7) / 8, B, 0, stream>>>(out, n);
  (void)n_in; (void)out_size; (void)ws_size;
}
